// Contextual_Attention_28793460752587
// MI455X (gfx1250) — compile-verified, matrix operand order checked by reading, not run on hardware
//
#include <hip/hip_runtime.h>
#include <hip/hip_bf16.h>
#include <math.h>

// ---------------------------------------------------------------------------
// Contextual attention for MI455X (gfx1250, wave32, WMMA).
//   Q = evo @ q_w^T + q_b
//   K = im2col5(evo) @ Weff^T + biasK        (conv3/conv5 folded into Weff)
//   V = plm @ v_w^T + v_b
//   out = softmax(mask(Q K^T / sqrt(96))) @ V + V    (flash-style, no BxLxL)
// Matmuls on v_wmma_f32_16x16x32_bf16; V staged to LDS with
// global_load_async_to_lds_b128 (ASYNCcnt) overlapping the QK^T WMMAs.
// ---------------------------------------------------------------------------

#define B_   8
#define L_   2048
#define QIN  512
#define VIN  1024
#define QK_  96
#define CATD 704
#define KEFF 2560   // 5 taps * 512 channels

typedef __attribute__((ext_vector_type(16))) __bf16 v16bf;
typedef __attribute__((ext_vector_type(8)))  __bf16 v8bf;
typedef __attribute__((ext_vector_type(8)))  float  v8f;

// ---------------------------------------------------------------------------
// Kernel 1: convert q_w / v_w to bf16.
// ---------------------------------------------------------------------------
__global__ void convert_w_kernel(const float* __restrict__ q_w,
                                 const float* __restrict__ v_w,
                                 __bf16* __restrict__ Wq,
                                 __bf16* __restrict__ Wv) {
  int i = blockIdx.x * 256 + threadIdx.x;
  if (i < QK_ * QIN) Wq[i] = (__bf16)q_w[i];
  if (i < QK_ * VIN) Wv[i] = (__bf16)v_w[i];
}

// ---------------------------------------------------------------------------
// Kernel 2: compose conv weights into a single 5-tap effective K-projection.
//   Weff[o, s*512+c] =  (s==2 ? k_w[o,c] : 0)
//                     + sum_o' k_w[o,512+o'] * cn3_w[o',c,s-1]   (s in 1..3)
//                     + sum_o' k_w[o,608+o'] * cn5_w[o',c,s]
//   biasK[o] = k_b[o] + k_w[o,512:608]·cn3_b + k_w[o,608:704]·cn5_b
// ---------------------------------------------------------------------------
__global__ void build_weff_kernel(const float* __restrict__ k_w,
                                  const float* __restrict__ k_b,
                                  const float* __restrict__ cn3_w,
                                  const float* __restrict__ cn3_b,
                                  const float* __restrict__ cn5_w,
                                  const float* __restrict__ cn5_b,
                                  __bf16* __restrict__ Weff,
                                  float* __restrict__ biasK) {
  int idx = blockIdx.x * 256 + threadIdx.x;
  if (idx < QK_ * KEFF) {
    int o   = idx / KEFF;
    int rem = idx - o * KEFF;
    int s   = rem / QIN;
    int c   = rem - s * QIN;
    const float* kw3 = k_w + (size_t)o * CATD + 512;
    const float* kw5 = k_w + (size_t)o * CATD + 608;
    float w = (s == 2) ? k_w[(size_t)o * CATD + c] : 0.0f;
    if (s >= 1 && s <= 3) {
      int t = s - 1;
      for (int op = 0; op < QK_; ++op)
        w += kw3[op] * cn3_w[((size_t)op * QIN + c) * 3 + t];
    }
    for (int op = 0; op < QK_; ++op)
      w += kw5[op] * cn5_w[((size_t)op * QIN + c) * 5 + s];
    Weff[(size_t)o * KEFF + rem] = (__bf16)w;
  }
  if (blockIdx.x == 0 && threadIdx.x < QK_) {
    int o = threadIdx.x;
    float bb = k_b[o];
    for (int op = 0; op < QK_; ++op) {
      bb += k_w[(size_t)o * CATD + 512 + op] * cn3_b[op];
      bb += k_w[(size_t)o * CATD + 608 + op] * cn5_b[op];
    }
    biasK[o] = bb;
  }
}

// ---------------------------------------------------------------------------
// Kernel 3: WMMA GEMM  C(M,96) = A(M,Kd) @ W^T + bias.
// One wave = 16 rows x 96 cols (6 accumulators). mode 1 = im2col over evo
// with 5 taps (K projection); modes 0/2 read A row-major directly.
// Fragment layout (ISA 7.12.2, 16-bit A 16x32): lane = M; elements 0..7 hold
// K = half*8..+7, elements 8..15 hold K = 16+half*8..+7.
// ---------------------------------------------------------------------------
__global__ __launch_bounds__(32) void qkv_gemm_kernel(
    const float* __restrict__ A, const __bf16* __restrict__ W,
    const float* __restrict__ bias, __bf16* __restrict__ Obf,
    float* __restrict__ Of32, int Kd, int mode) {
  const int lane = threadIdx.x;
  const int lo   = lane & 15;
  const int half = lane >> 4;
  const int mbase = blockIdx.x * 16;
  const int m = mbase + lo;
  const int b = m >> 11;          // L_ = 2048
  const int l = m & (L_ - 1);

  v8f acc[6] = {};

  for (int kc = 0; kc < Kd; kc += 32) {
    // ---- A fragment (f32 -> bf16 on the fly) ----
    v16bf af;
    if (mode != 1) {
      const float* p = A + (size_t)m * Kd;
      int c1 = kc + half * 8;
      #pragma unroll
      for (int i = 0; i < 8; ++i) {
        af[i]     = (__bf16)p[c1 + i];
        af[i + 8] = (__bf16)p[c1 + 16 + i];
      }
    } else {
      #pragma unroll
      for (int r = 0; r < 2; ++r) {
        int col = kc + half * 8 + r * 16;
        int s   = col >> 9;             // tap index 0..4
        int c   = col & (QIN - 1);
        int lr  = l + s - 2;            // SAME padding, zero OOB
        bool ok = (lr >= 0) && (lr < L_);
        const float* p = A + ((size_t)b * L_ + (ok ? lr : 0)) * QIN;
        #pragma unroll
        for (int i = 0; i < 8; ++i)
          af[r * 8 + i] = ok ? (__bf16)p[c + i] : (__bf16)0.0f;
      }
    }
    // ---- B fragments (weights already bf16) + WMMA ----
    #pragma unroll
    for (int t = 0; t < 6; ++t) {
      const __bf16* wp = W + (size_t)(t * 16 + lo) * Kd + kc + half * 8;
      const v8bf blo = *(const v8bf*)wp;
      const v8bf bhi = *(const v8bf*)(wp + 16);
      v16bf bf;
      #pragma unroll
      for (int i = 0; i < 8; ++i) { bf[i] = blo[i]; bf[i + 8] = bhi[i]; }
      acc[t] = __builtin_amdgcn_wmma_f32_16x16x32_bf16(
          false, af, false, bf, (short)0, acc[t], false, false);
    }
  }
  // ---- epilogue: bias, store bf16 (+ optional f32 for residual V) ----
  #pragma unroll
  for (int t = 0; t < 6; ++t) {
    int n = t * 16 + lo;
    float bv = bias[n];
    #pragma unroll
    for (int j = 0; j < 8; ++j) {
      int row = mbase + half * 8 + j;
      float v = acc[t][j] + bv;
      Obf[(size_t)row * QK_ + n] = (__bf16)v;
      if (Of32) Of32[(size_t)row * QK_ + n] = v;
    }
  }
}

// ---------------------------------------------------------------------------
// Kernel 4: flash attention. One wave per 16-query tile, online softmax,
// 32-key blocks. Score C-layout: VGPR j -> row 8*half+j, lane&15 -> key col.
// V blocks are staged to LDS with global_load_async_to_lds_b128 (ASYNCcnt),
// overlapping the QK^T WMMAs + softmax; s_wait_asynccnt 0 gates the PV gather.
// ---------------------------------------------------------------------------
__global__ __launch_bounds__(32) void attention_kernel(
    const __bf16* __restrict__ Qbf, const __bf16* __restrict__ Kbf,
    const __bf16* __restrict__ Vbf, const float* __restrict__ Vf32,
    const int* __restrict__ seqlengths, float* __restrict__ out) {
  __shared__ __align__(32) __bf16 Vlds[32 * 112];   // V block, key-major
  __shared__ __align__(32) __bf16 Plds[16 * 48];    // P transpose staging

  const int lane = threadIdx.x;
  const int lo   = lane & 15;
  const int half = lane >> 4;
  const int b     = blockIdx.x >> 7;          // L_/16 = 128 tiles
  const int qbase = (blockIdx.x & 127) * 16;
  const int seqlen = seqlengths[b];
  const float norm = 0.10206207261596577f;    // 1/sqrt(96)

  // LDS byte offset of Vlds (generic LDS address keeps offset in addr[31:0])
  const unsigned vlds_base = (unsigned)(unsigned long long)(void*)&Vlds[0];

  // Q fragments (row = qbase+lo), 3 chunks of d=32
  const __bf16* qrow = Qbf + ((size_t)b * L_ + qbase + lo) * QK_;
  v16bf qf[3];
  #pragma unroll
  for (int c = 0; c < 3; ++c) {
    int c1 = c * 32 + half * 8;
    const v8bf a0 = *(const v8bf*)(qrow + c1);
    const v8bf a1 = *(const v8bf*)(qrow + c1 + 16);
    #pragma unroll
    for (int i = 0; i < 8; ++i) { qf[c][i] = a0[i]; qf[c][i + 8] = a1[i]; }
  }

  v8f oacc[6] = {};
  float mrun[8], lrun[8];
  #pragma unroll
  for (int j = 0; j < 8; ++j) { mrun[j] = -3.0e38f; lrun[j] = 0.0f; }

  for (int kb = 0; kb < L_; kb += 32) {
    // ---- async-stage V block: 32 rows x 192B = 384 x b128 chunks ----
    // lane-parallel: 12 issues/lane-loop, each lane moves one 16B chunk.
    #pragma unroll
    for (int it = 0; it < 12; ++it) {
      int chunk = it * 32 + lane;        // 0..383
      int row   = chunk / 12;            // key row in block (0..31)
      int part  = chunk - row * 12;      // 16B chunk within row (0..11)
      unsigned lds_off = vlds_base + (unsigned)(row * 224 + part * 16);
      const __bf16* gp = Vbf + ((size_t)b * L_ + kb + row) * QK_ + part * 8;
      asm volatile("global_load_async_to_lds_b128 %0, %1, off"
                   :: "v"(lds_off), "v"(gp)
                   : "memory");
    }
    if (kb + 32 < L_)  // pull next K block toward the WGP (global_prefetch_b8)
      __builtin_prefetch(Kbf + ((size_t)b * L_ + kb + 32 + lane) * QK_, 0, 1);

    // ---- S = Q K^T for two 16-key subtiles (overlaps async V copy) ----
    v8f s0 = {}, s1 = {};
    #pragma unroll
    for (int c = 0; c < 3; ++c) {
      const __bf16* k0 = Kbf + ((size_t)b * L_ + kb + lo) * QK_ + c * 32 + half * 8;
      const __bf16* k1 = k0 + 16 * QK_;
      v16bf bf0, bf1;
      { const v8bf x0 = *(const v8bf*)k0, x1 = *(const v8bf*)(k0 + 16);
        #pragma unroll
        for (int i = 0; i < 8; ++i) { bf0[i] = x0[i]; bf0[i + 8] = x1[i]; } }
      { const v8bf x0 = *(const v8bf*)k1, x1 = *(const v8bf*)(k1 + 16);
        #pragma unroll
        for (int i = 0; i < 8; ++i) { bf1[i] = x0[i]; bf1[i + 8] = x1[i]; } }
      s0 = __builtin_amdgcn_wmma_f32_16x16x32_bf16(false, qf[c], false, bf0,
                                                   (short)0, s0, false, false);
      s1 = __builtin_amdgcn_wmma_f32_16x16x32_bf16(false, qf[c], false, bf1,
                                                   (short)0, s1, false, false);
    }
    // ---- scale + key mask (this lane's key column) ----
    const bool mk0 = (kb + lo) >= seqlen;
    const bool mk1 = (kb + 16 + lo) >= seqlen;
    float a0[8], a1[8];
    #pragma unroll
    for (int j = 0; j < 8; ++j) {
      a0[j] = mk0 ? -1.0e6f : s0[j] * norm;
      a1[j] = mk1 ? -1.0e6f : s1[j] * norm;
    }
    __syncthreads();   // Plds free for reuse
    // ---- online softmax update, write P to LDS in row-major ----
    #pragma unroll
    for (int j = 0; j < 8; ++j) {
      float rmax = fmaxf(a0[j], a1[j]);
      #pragma unroll
      for (int d = 1; d < 16; d <<= 1)
        rmax = fmaxf(rmax, __shfl_xor(rmax, d, 32));
      float mnew  = fmaxf(mrun[j], rmax);
      float alpha = __expf(mrun[j] - mnew);
      float p0 = __expf(a0[j] - mnew);
      float p1 = __expf(a1[j] - mnew);
      float rsum = p0 + p1;
      #pragma unroll
      for (int d = 1; d < 16; d <<= 1)
        rsum += __shfl_xor(rsum, d, 32);
      lrun[j] = lrun[j] * alpha + rsum;
      mrun[j] = mnew;
      #pragma unroll
      for (int t = 0; t < 6; ++t) oacc[t][j] *= alpha;   // rescale O row
      int prow = 8 * half + j;
      Plds[prow * 48 + lo]      = (__bf16)p0;
      Plds[prow * 48 + 16 + lo] = (__bf16)p1;
    }
    __syncthreads();
    // ---- wait for async V copy, then O += P V ----
    asm volatile("s_wait_asynccnt 0x0" ::: "memory");
    v16bf pf;
    {
      const v8bf p0v = *(const v8bf*)(Plds + lo * 48 + half * 8);
      const v8bf p1v = *(const v8bf*)(Plds + lo * 48 + 16 + half * 8);
      #pragma unroll
      for (int i = 0; i < 8; ++i) { pf[i] = p0v[i]; pf[i + 8] = p1v[i]; }
    }
    #pragma unroll
    for (int t = 0; t < 6; ++t) {
      int d = t * 16 + lo;
      v16bf vf;
      #pragma unroll
      for (int i = 0; i < 8; ++i) {
        vf[i]     = Vlds[(half * 8 + i) * 112 + d];
        vf[i + 8] = Vlds[(16 + half * 8 + i) * 112 + d];
      }
      oacc[t] = __builtin_amdgcn_wmma_f32_16x16x32_bf16(
          false, pf, false, vf, (short)0, oacc[t], false, false);
    }
    __syncthreads();   // protect Vlds before next block's async staging
  }
  // ---- epilogue: normalize rows, add V residual ----
  #pragma unroll
  for (int t = 0; t < 6; ++t) {
    int d = t * 16 + lo;
    #pragma unroll
    for (int j = 0; j < 8; ++j) {
      int row = qbase + 8 * half + j;
      size_t idx = ((size_t)b * L_ + row) * QK_ + d;
      out[idx] = oacc[t][j] / lrun[j] + Vf32[idx];
    }
  }
}

// ---------------------------------------------------------------------------
extern "C" void kernel_launch(void* const* d_in, const int* in_sizes, int n_in,
                              void* d_out, int out_size, void* d_ws,
                              size_t ws_size, hipStream_t stream) {
  (void)in_sizes; (void)n_in; (void)out_size; (void)ws_size;
  const float* plm   = (const float*)d_in[0];
  const float* evo   = (const float*)d_in[1];
  const int*   seql  = (const int*)d_in[2];
  const float* q_w   = (const float*)d_in[3];
  const float* q_b   = (const float*)d_in[4];
  const float* k_w   = (const float*)d_in[5];
  const float* k_b   = (const float*)d_in[6];
  const float* v_w   = (const float*)d_in[7];
  const float* v_b   = (const float*)d_in[8];
  const float* cn3_w = (const float*)d_in[9];
  const float* cn3_b = (const float*)d_in[10];
  const float* cn5_w = (const float*)d_in[11];
  const float* cn5_b = (const float*)d_in[12];
  float* out = (float*)d_out;

  // workspace carve-out (256B aligned), ~16.6 MB total
  size_t off = 0;
  auto take = [&](size_t bytes) {
    char* p = (char*)d_ws + off;
    off += (bytes + 255) & ~(size_t)255;
    return (void*)p;
  };
  const size_t ML = (size_t)B_ * L_;
  __bf16* Wq    = (__bf16*)take((size_t)QK_ * QIN * 2);
  __bf16* Wv    = (__bf16*)take((size_t)QK_ * VIN * 2);
  __bf16* Weff  = (__bf16*)take((size_t)QK_ * KEFF * 2);
  float*  biasK = (float*)take((size_t)QK_ * 4);
  __bf16* Qbf   = (__bf16*)take(ML * QK_ * 2);
  __bf16* Kbf   = (__bf16*)take(ML * QK_ * 2);
  __bf16* Vbf   = (__bf16*)take(ML * QK_ * 2);
  float*  Vf32  = (float*)take(ML * QK_ * 4);

  convert_w_kernel<<<384, 256, 0, stream>>>(q_w, v_w, Wq, Wv);
  build_weff_kernel<<<960, 256, 0, stream>>>(k_w, k_b, cn3_w, cn3_b, cn5_w,
                                             cn5_b, Weff, biasK);
  const int gemm_blocks = (int)(ML / 16);  // 1024
  qkv_gemm_kernel<<<gemm_blocks, 32, 0, stream>>>(evo, Wq, q_b, Qbf, nullptr,
                                                  QIN, 0);
  qkv_gemm_kernel<<<gemm_blocks, 32, 0, stream>>>(evo, Weff, biasK, Kbf,
                                                  nullptr, KEFF, 1);
  qkv_gemm_kernel<<<gemm_blocks, 32, 0, stream>>>(plm, Wv, v_b, Vbf, Vf32,
                                                  VIN, 2);
  attention_kernel<<<B_ * (L_ / 16), 32, 0, stream>>>(Qbf, Kbf, Vbf, Vf32,
                                                      seql, out);
}
